// SoftTimeAttention_28003186769980
// MI455X (gfx1250) — compile-verified, matrix operand order checked by reading, not run on hardware
//
#include <hip/hip_runtime.h>
#include <math.h>

typedef __attribute__((ext_vector_type(16))) __bf16 v16bf;
typedef __attribute__((ext_vector_type(8)))  __bf16 v8bf;
typedef __attribute__((ext_vector_type(8)))  float  v8f;

#define T_SEQ 4096
#define H_DIM 256
#define B_DIM 4
#define KC    64          // keys per block = 2 WMMA K-steps
#define QB    64          // queries per workgroup (4 qtiles x 16)
#define NT    8           // 8 N-tiles of 16 cols = 128 H per wave
#define KPAD  72          // LDS row stride in halfs (multiple of 8 -> 16B aligned)
// T * sqrt(log2(e)) : puts logits directly in the exp2 domain
#define TSCALE 4919.7974f

__global__ __launch_bounds__(256)
void soft_time_attn_kernel(const float* __restrict__ h_seq,
                           const float* __restrict__ t,
                           float* __restrict__ out)
{
    __shared__ __bf16 lds_v[H_DIM * KPAD];   // V tile, transposed: [h][k] bf16
    __shared__ float  lds_t[KC];             // pre-scaled key times

    const int tid   = threadIdx.x;
    const int lane  = tid & 31;
    const int wave  = tid >> 5;
    const int qtile = wave >> 1;             // 0..3 -> 16-query tile
    const int hhalf = wave & 1;              // 0..1 -> 128-col H half
    const int b     = blockIdx.x / (T_SEQ / QB);
    const int q0    = (blockIdx.x % (T_SEQ / QB)) * QB + qtile * 16;
    const int row   = lane & 15;             // M (A row) == N (B col) index
    const int hi    = lane >> 4;
    const int h0    = hhalf * (NT * 16);

    const float tqs = TSCALE * t[b * T_SEQ + q0 + row];

    float m = -INFINITY;
    float l = 0.0f;
    v8f c[NT];
    #pragma unroll
    for (int n = 0; n < NT; ++n)
        #pragma unroll
        for (int r = 0; r < 8; ++r) c[n][r] = 0.0f;

    // ---- staging helpers: group g = (k8, h4) covers rows k8*8..+7 at float4 col h4
    auto load_group = [&](int g, float4* dst, const float* hbase) {
        int h4 = g & 63, k8 = g >> 6;
        const float4* s = (const float4*)hbase + (size_t)(k8 * 8) * (H_DIM / 4) + h4;
        #pragma unroll
        for (int j = 0; j < 8; ++j) dst[j] = s[j * (H_DIM / 4)];
    };
    auto store_group = [&](int g, const float4* v) {
        int h4 = g & 63, k8 = g >> 6;
        #pragma unroll
        for (int comp = 0; comp < 4; ++comp) {
            v8bf pk;
            #pragma unroll
            for (int j = 0; j < 8; ++j) {
                const float* f = (const float*)&v[j];
                pk[j] = (__bf16)f[comp];
            }
            *(v8bf*)&lds_v[(h4 * 4 + comp) * KPAD + k8 * 8] = pk;
        }
    };

    // prefetch block 0
    float4 pre[2][8];
    float  pret = 0.0f;
    {
        const float* hbase = h_seq + (size_t)(b * T_SEQ) * H_DIM;
        load_group(tid,       pre[0], hbase);
        load_group(tid + 256, pre[1], hbase);
        if (tid < KC) pret = TSCALE * t[b * T_SEQ + tid];
    }

    for (int k0 = 0; k0 < T_SEQ; k0 += KC) {
        // ---- drain prefetch registers into LDS (cvt f32 -> bf16, transposed)
        store_group(tid,       pre[0]);
        store_group(tid + 256, pre[1]);
        if (tid < KC) lds_t[tid] = pret;
        __syncthreads();

        // ---- issue next block's global loads (overlaps compute below)
        if (k0 + KC < T_SEQ) {
            const float* hbase = h_seq + (size_t)(b * T_SEQ + k0 + KC) * H_DIM;
            load_group(tid,       pre[0], hbase);
            load_group(tid + 256, pre[1], hbase);
            if (tid < KC) pret = TSCALE * t[b * T_SEQ + k0 + KC + tid];
        }

        // ---- key times for this lane's 32 fragment elements (8x ds_load_b128 worth)
        float tkv[2][16];
        #pragma unroll
        for (int ch = 0; ch < 2; ++ch) {
            const float* tb = &lds_t[ch * 32 + hi * 8];
            float4 A = *(const float4*)(tb + 0);
            float4 Bv = *(const float4*)(tb + 4);
            float4 Cv = *(const float4*)(tb + 16);
            float4 Dv = *(const float4*)(tb + 20);
            tkv[ch][0]=A.x;  tkv[ch][1]=A.y;  tkv[ch][2]=A.z;  tkv[ch][3]=A.w;
            tkv[ch][4]=Bv.x; tkv[ch][5]=Bv.y; tkv[ch][6]=Bv.z; tkv[ch][7]=Bv.w;
            tkv[ch][8]=Cv.x; tkv[ch][9]=Cv.y; tkv[ch][10]=Cv.z; tkv[ch][11]=Cv.w;
            tkv[ch][12]=Dv.x; tkv[ch][13]=Dv.y; tkv[ch][14]=Dv.z; tkv[ch][15]=Dv.w;
        }

        // ---- logits (log2 domain) in A-fragment order; K = (e<8?e:e+8)+8*hi
        float p[2][16];
        float smax = -INFINITY;
        #pragma unroll
        for (int ch = 0; ch < 2; ++ch)
            #pragma unroll
            for (int e = 0; e < 16; ++e) {
                float dt = tqs - tkv[ch][e];
                float s  = -(dt * dt);
                p[ch][e] = s;
                smax = fmaxf(smax, s);
            }
        smax = fmaxf(smax, __shfl_xor(smax, 16, 32));
        float m_new = fmaxf(m, smax);
        float scale = exp2f(m - m_new);      // 1.0 when row max unchanged

        float rsum = 0.0f;
        v16bf a[2];
        #pragma unroll
        for (int ch = 0; ch < 2; ++ch)
            #pragma unroll
            for (int e = 0; e < 16; ++e) {
                float pe = exp2f(p[ch][e] - m_new);
                rsum += pe;
                a[ch][e] = (__bf16)pe;
            }
        rsum += __shfl_xor(rsum, 16, 32);
        l = l * scale + rsum;
        m = m_new;

        // ---- rescale accumulators only when some row's max actually moved
        if (__ballot(scale < 1.0f)) {
            float sc[8];
            #pragma unroll
            for (int r = 0; r < 8; ++r) sc[r] = __shfl(scale, r + 8 * hi, 32);
            #pragma unroll
            for (int n = 0; n < NT; ++n)
                #pragma unroll
                for (int r = 0; r < 8; ++r) c[n][r] *= sc[r];
        }

        // ---- P @ V : 16x v_wmma_f32_16x16x32_bf16 per key block
        #pragma unroll
        for (int n = 0; n < NT; ++n) {
            const __bf16* vr = &lds_v[(h0 + n * 16 + row) * KPAD];
            v8bf b0lo = *(const v8bf*)(vr + hi * 8);
            v8bf b0hi = *(const v8bf*)(vr + 16 + hi * 8);
            v8bf b1lo = *(const v8bf*)(vr + 32 + hi * 8);
            v8bf b1hi = *(const v8bf*)(vr + 48 + hi * 8);
            v16bf bf0, bf1;
            #pragma unroll
            for (int e = 0; e < 8; ++e) {
                bf0[e] = b0lo[e]; bf0[e + 8] = b0hi[e];
                bf1[e] = b1lo[e]; bf1[e + 8] = b1hi[e];
            }
            c[n] = __builtin_amdgcn_wmma_f32_16x16x32_bf16(
                false, a[0], false, bf0, (short)0, c[n], false, false);
            c[n] = __builtin_amdgcn_wmma_f32_16x16x32_bf16(
                false, a[1], false, bf1, (short)0, c[n], false, false);
        }
        __syncthreads();
    }

    // ---- epilogue: divide by softmax denominator, write f32
    float invl[8];
    #pragma unroll
    for (int r = 0; r < 8; ++r) invl[r] = 1.0f / __shfl(l, r + 8 * hi, 32);

    #pragma unroll
    for (int n = 0; n < NT; ++n) {
        int col = h0 + n * 16 + row;
        #pragma unroll
        for (int r = 0; r < 8; ++r) {
            int qr = q0 + r + 8 * hi;
            out[(size_t)(b * T_SEQ + qr) * H_DIM + col] = c[n][r] * invl[r];
        }
    }
}

extern "C" void kernel_launch(void* const* d_in, const int* in_sizes, int n_in,
                              void* d_out, int out_size, void* d_ws, size_t ws_size,
                              hipStream_t stream) {
    const float* h_seq = (const float*)d_in[0];   // (B, T, H) f32
    const float* t     = (const float*)d_in[1];   // (B, T, 1) f32
    float*       out   = (float*)d_out;           // (B, T, H) f32
    (void)in_sizes; (void)n_in; (void)out_size; (void)d_ws; (void)ws_size;

    dim3 grid(B_DIM * (T_SEQ / QB));   // 256 workgroups
    dim3 block(256);                   // 8 wave32s
    soft_time_attn_kernel<<<grid, block, 0, stream>>>(h_seq, t, out);
}
